// Semantic_Wise_Attention_51651276701961
// MI455X (gfx1250) — matrix operand order repaired and checked by reading, not compiled, not run on hardware
//
#include <hip/hip_runtime.h>
#include <hip/hip_bf16.h>

// ---------------------------------------------------------------------------
// Problem constants (from reference)
#define Bdim 8
#define Tdim 256
#define Ldim 48
#define FDdim 1024
#define WDdim 512
#define Hdim 512
#define Odim 512

typedef __attribute__((ext_vector_type(16))) __bf16 v16bf;
typedef __attribute__((ext_vector_type(8)))  __bf16 v8bf;
typedef __attribute__((ext_vector_type(8)))  float  v8f;

__device__ __forceinline__ v16bf cat8(v8bf lo, v8bf hi) {
  return __builtin_shufflevector(lo, hi, 0,1,2,3,4,5,6,7,8,9,10,11,12,13,14,15);
}

// A-fragment (16x32 bf16, row-major source, ld elements).
// Lane layout: m = lane&15, half = lane>>4,
//   elements 0..15 = A[m][16*half + 0..15]
__device__ __forceinline__ v16bf load_a_frag(const __bf16* __restrict__ base,
                                             int ld, int lane) {
  int m = lane & 15, half = lane >> 4;
  const __bf16* p = base + (long)m * ld + 16 * half;
  return *(const v16bf*)(p);
}

// B-fragment (32x16 bf16) from B^T stored row-major [N][K]:
//   col = lane&15, half = lane>>4, elements = BT[col][16*half + 0..15]
__device__ __forceinline__ v16bf load_bt_frag(const __bf16* __restrict__ base,
                                              int ld, int lane) {
  int col = lane & 15, half = lane >> 4;
  return *(const v16bf*)(base + (long)col * ld + 16 * half);
}

__device__ __forceinline__ v8f wmma_bf16(v16bf a, v16bf b, v8f c) {
  return __builtin_amdgcn_wmma_f32_16x16x32_bf16(false, a, false, b,
                                                 (short)0, c, false, false);
}

// ---------------------------------------------------------------------------
// Prep kernels: f32 -> bf16 copy, and f32 -> bf16 transpose (dst[cols][rows])
__global__ void k_cvt_bf16(const float* __restrict__ src, __bf16* __restrict__ dst, int n) {
  int i = blockIdx.x * blockDim.x + threadIdx.x;
  if (i < n) dst[i] = (__bf16)src[i];
}

__global__ void k_transpose_bf16(const float* __restrict__ src, __bf16* __restrict__ dst,
                                 int rows, int cols) {
  int i = blockIdx.x * blockDim.x + threadIdx.x;
  if (i < rows * cols) {
    int r = i / cols, c = i - r * cols;
    dst[(long)c * rows + r] = (__bf16)src[i];
  }
}

// ---------------------------------------------------------------------------
// Generic bf16 WMMA GEMM: C[M][N] = act(A[M][K] * BT[N][K]^T + bias)
// One wave per 64x16 output block (4 m-tiles share one B fragment per k-step:
// 4 independent WMMA accumulator chains, 4x less B traffic).
// Requires M % 64 == 0, N % 16 == 0, K % 32 == 0.
__global__ __launch_bounds__(256)
void k_gemm_bf16(const __bf16* __restrict__ A, const __bf16* __restrict__ BT,
                 const float* __restrict__ bias, float* __restrict__ C,
                 int M, int N, int K, int do_relu) {
  int tilesN = N >> 4;
  int tilesTotal = (M >> 6) * tilesN;      // 64-row groups
  int nWaves = (gridDim.x * blockDim.x) >> 5;
  int waveId = (blockIdx.x * blockDim.x + threadIdx.x) >> 5;
  int lane = threadIdx.x & 31;
  int col = lane & 15, half = lane >> 4;

  for (int tile = waveId; tile < tilesTotal; tile += nWaves) {
    int tg = tile / tilesN, tn = tile - tg * tilesN;
    int m0 = tg * 64;
    const __bf16* A0 = A + (long)m0 * K;
    const __bf16* A1 = A0 + (long)16 * K;
    const __bf16* A2 = A0 + (long)32 * K;
    const __bf16* A3 = A0 + (long)48 * K;
    const __bf16* Bbase = BT + (long)(tn * 16) * K;
    v8f acc0 = {}, acc1 = {}, acc2 = {}, acc3 = {};
    for (int k0 = 0; k0 < K; k0 += 32) {
      v16bf b  = load_bt_frag(Bbase + k0, K, lane);
      v16bf a0 = load_a_frag(A0 + k0, K, lane);
      v16bf a1 = load_a_frag(A1 + k0, K, lane);
      v16bf a2 = load_a_frag(A2 + k0, K, lane);
      v16bf a3 = load_a_frag(A3 + k0, K, lane);
      acc0 = wmma_bf16(a0, b, acc0);
      acc1 = wmma_bf16(a1, b, acc1);
      acc2 = wmma_bf16(a2, b, acc2);
      acc3 = wmma_bf16(a3, b, acc3);
    }
    int n = tn * 16 + col;
    float bv = bias ? bias[n] : 0.0f;
    #pragma unroll
    for (int r = 0; r < 8; ++r) {
      int mrow = m0 + r + 8 * half;
      float x0 = acc0[r] + bv;
      float x1 = acc1[r] + bv;
      float x2 = acc2[r] + bv;
      float x3 = acc3[r] + bv;
      if (do_relu) {
        x0 = fmaxf(x0, 0.0f); x1 = fmaxf(x1, 0.0f);
        x2 = fmaxf(x2, 0.0f); x3 = fmaxf(x3, 0.0f);
      }
      C[(long)(mrow)      * N + n] = x0;
      C[(long)(mrow + 16) * N + n] = x1;
      C[(long)(mrow + 32) * N + n] = x2;
      C[(long)(mrow + 48) * N + n] = x3;
    }
  }
}

// ---------------------------------------------------------------------------
// Fused attention kernel. One 256-thread block per (b,t).
//   sV        = v[b,t,:] staged once                         (512 f32, 2 KB)
//   sA[l][k]  = bf16 tanh(v[b,t,k] + h[b,l,k])               (48x512 bf16, 48 KB)
//   two passes over n-halves of 256 columns:
//     sR[l][c] = relu( (sA * WaT_half)[l][c] )               (48x256 f32, 48 KB)
//     per-column masked softmax over l, write h2v, accumulate fww
// GEMM: each wave owns n-tiles; 3 l-tiles share the B fragment (3 independent
// accumulator chains -> no WMMA RAW stalls, 3x less Wa traffic).
__global__ __launch_bounds__(256)
void k_attn_fused(const float* __restrict__ venc,   // [B*T][H]
                  const float* __restrict__ henc,   // [B*L][H]
                  const float* __restrict__ wordf,  // [B][L][WD]
                  const __bf16* __restrict__ WaT,   // [H][H] (transposed, bf16)
                  const int* __restrict__ flen, const int* __restrict__ wlen,
                  float* __restrict__ h2v_out,      // [B*T][L][H]
                  __bf16* __restrict__ fww_bf) {    // [B*T][H]
  __shared__ float  sV[Hdim];
  __shared__ __bf16 sA[Ldim * Hdim];
  __shared__ float  sR[Ldim * 256];

  int bt = blockIdx.x;
  int b = bt >> 8;          // T == 256
  int t = bt & (Tdim - 1);
  int tid = threadIdx.x;

  const float* vrow = venc + (long)bt * Hdim;
  const float* hb   = henc + (long)b * Ldim * Hdim;

  // Stage v row once
  sV[tid] = vrow[tid];
  sV[tid + 256] = vrow[tid + 256];
  __syncthreads();

  // Stage A: tanh(v + h) -> bf16 LDS
  for (int idx = tid; idx < Ldim * Hdim; idx += 256) {
    int l = idx >> 9, k = idx & (Hdim - 1);
    sA[idx] = (__bf16)tanhf(sV[k] + hb[l * Hdim + k]);
  }
  __syncthreads();

  int wave = tid >> 5, lane = tid & 31;
  int col = lane & 15, half = lane >> 4;
  int fl = flen[b], wl = wlen[b];
  bool vmask = (t < fl);
  float* h2v_bt = h2v_out + (long)bt * Ldim * Hdim;

  for (int npass = 0; npass < 2; ++npass) {
    int nbase = npass * 256;

    // GEMM: 16 n-tiles over 8 waves; 3 l-tiles per n-tile in registers
    for (int nt = wave; nt < 16; nt += 8) {
      const __bf16* Bbase = WaT + (long)(nbase + nt * 16) * Hdim;
      v8f acc0 = {}, acc1 = {}, acc2 = {};
      for (int k0 = 0; k0 < Hdim; k0 += 32) {
        v16bf bb = load_bt_frag(Bbase + k0, Hdim, lane);
        v16bf a0 = load_a_frag(sA + k0, Hdim, lane);
        v16bf a1 = load_a_frag(sA + (long)16 * Hdim + k0, Hdim, lane);
        v16bf a2 = load_a_frag(sA + (long)32 * Hdim + k0, Hdim, lane);
        acc0 = wmma_bf16(a0, bb, acc0);
        acc1 = wmma_bf16(a1, bb, acc1);
        acc2 = wmma_bf16(a2, bb, acc2);
      }
      int c = nt * 16 + col;
      #pragma unroll
      for (int r = 0; r < 8; ++r) {
        int lr = r + 8 * half;
        sR[(lr)      * 256 + c] = fmaxf(acc0[r], 0.0f);
        sR[(lr + 16) * 256 + c] = fmaxf(acc1[r], 0.0f);
        sR[(lr + 32) * 256 + c] = fmaxf(acc2[r], 0.0f);
      }
    }
    __syncthreads();

    // Masked softmax over l + weighted reduce; one column per thread
    {
      int c = tid;               // local column
      int hcol = nbase + c;      // global hidden index
      float mx = -3.0e38f;
      #pragma unroll 4
      for (int l = 0; l < Ldim; ++l) {
        float x = (vmask && l < wl) ? sR[l * 256 + c] : -1.0e32f;
        mx = fmaxf(mx, x);
      }
      float sum = 0.0f;
      #pragma unroll 4
      for (int l = 0; l < Ldim; ++l) {
        float x = (vmask && l < wl) ? sR[l * 256 + c] : -1.0e32f;
        float e = expf(x - mx);
        sR[l * 256 + c] = e;
        sum += e;
      }
      float inv = 1.0f / sum;
      float fww = 0.0f;
      #pragma unroll 4
      for (int l = 0; l < Ldim; ++l) {
        float p = sR[l * 256 + c] * inv;
        p = (vmask && l < wl) ? p : 0.0f;
        h2v_bt[l * Hdim + hcol] = p;
        fww += p * wordf[((long)b * Ldim + l) * WDdim + hcol];
      }
      fww_bf[(long)bt * Hdim + hcol] = (__bf16)fww;
    }
    __syncthreads();  // sR reused next pass
  }
}

// ---------------------------------------------------------------------------
extern "C" void kernel_launch(void* const* d_in, const int* in_sizes, int n_in,
                              void* d_out, int out_size, void* d_ws, size_t ws_size,
                              hipStream_t stream) {
  (void)in_sizes; (void)n_in; (void)out_size; (void)ws_size;

  const float* frame = (const float*)d_in[0];  // [B,T,FD]
  const float* word  = (const float*)d_in[1];  // [B,L,WD]
  const int*   flen  = (const int*)d_in[2];    // [B]
  const int*   wlen  = (const int*)d_in[3];    // [B]
  const float* Wf    = (const float*)d_in[4];  // [FD,H]
  const float* bf_   = (const float*)d_in[5];  // [H]
  const float* Ww    = (const float*)d_in[6];  // [WD,H]
  const float* bw    = (const float*)d_in[7];  // [H]
  const float* Wa    = (const float*)d_in[8];  // [H,H]
  const float* Wout  = (const float*)d_in[9];  // [H,O]

  // Output: out [B,T,O] then h2v [B,T,L,H], concatenated f32
  float* out = (float*)d_out;
  float* h2v = out + (size_t)Bdim * Tdim * Odim;

  // Workspace layout (bytes, 256B-aligned offsets)
  char* ws = (char*)d_ws;
  __bf16* frame_bf = (__bf16*)(ws + 0);                 // 2048*1024*2 = 4 MiB
  __bf16* WfT_bf   = (__bf16*)(ws + 4194304);           // 512*1024*2  = 1 MiB
  __bf16* word_bf  = (__bf16*)(ws + 5242880);           // 384*512*2
  __bf16* WwT_bf   = (__bf16*)(ws + 5636096);           // 512*512*2
  __bf16* WaT_bf   = (__bf16*)(ws + 6160384);           // 512*512*2
  __bf16* WoutT_bf = (__bf16*)(ws + 6684672);           // 512*512*2
  float*  v_f32    = (float*)(ws + 7208960);            // 2048*512*4  = 4 MiB
  float*  h_f32    = (float*)(ws + 11403264);           // 384*512*4
  __bf16* fww_bf   = (__bf16*)(ws + 12189696);          // 2048*512*2 -> end ~14.3 MiB

  const int BT = Bdim * Tdim;   // 2048
  const int BL = Bdim * Ldim;   // 384

  // --- prep: bf16 conversions / weight transposes ---
  {
    int n = BT * FDdim;
    k_cvt_bf16<<<(n + 255) / 256, 256, 0, stream>>>(frame, frame_bf, n);
  }
  {
    int n = BL * WDdim;
    k_cvt_bf16<<<(n + 255) / 256, 256, 0, stream>>>(word, word_bf, n);
  }
  k_transpose_bf16<<<(FDdim * Hdim + 255) / 256, 256, 0, stream>>>(Wf, WfT_bf, FDdim, Hdim);
  k_transpose_bf16<<<(WDdim * Hdim + 255) / 256, 256, 0, stream>>>(Ww, WwT_bf, WDdim, Hdim);
  k_transpose_bf16<<<(Hdim * Hdim + 255) / 256, 256, 0, stream>>>(Wa, WaT_bf, Hdim, Hdim);
  k_transpose_bf16<<<(Hdim * Odim + 255) / 256, 256, 0, stream>>>(Wout, WoutT_bf, Hdim, Odim);

  // --- encoders ---
  {
    int tiles = (BT / 64) * (Hdim / 16);              // 1024 wave-blocks
    k_gemm_bf16<<<tiles / 8, 256, 0, stream>>>(frame_bf, WfT_bf, bf_, v_f32,
                                               BT, Hdim, FDdim, 1);
  }
  {
    int tiles = (BL / 64) * (Hdim / 16);              // 192 wave-blocks
    k_gemm_bf16<<<tiles / 8, 256, 0, stream>>>(word_bf, WwT_bf, bw, h_f32,
                                               BL, Hdim, WDdim, 1);
  }

  // --- fused additive attention + softmax + weighted sum ---
  k_attn_fused<<<BT, 256, 0, stream>>>(v_f32, h_f32, word, WaT_bf,
                                       flen, wlen, h2v, fww_bf);

  // --- final projection: out = fww @ Wout ---
  {
    int tiles = (BT / 64) * (Odim / 16);              // 1024 wave-blocks
    k_gemm_bf16<<<tiles / 8, 256, 0, stream>>>(fww_bf, WoutT_bf, nullptr, out,
                                               BT, Odim, Hdim, 0);
  }
}
